// Deepset_44117904065272
// MI455X (gfx1250) — matrix operand order repaired and checked by reading, not compiled, not run on hardware
//
#include <hip/hip_runtime.h>
#include <stdint.h>

typedef _Float16 v16h __attribute__((ext_vector_type(16)));
typedef _Float16 v8h  __attribute__((ext_vector_type(8)));
typedef float    v8f  __attribute__((ext_vector_type(8)));
typedef float    f32x4 __attribute__((ext_vector_type(4)));

#define DIM    128
#define MID_E  96
#define HID    64
#define LN_EPS 1e-5f

__device__ __forceinline__ float fast_tanh(float v) {
#if __has_builtin(__builtin_amdgcn_tanhf)
  return __builtin_amdgcn_tanhf(v);
#elif __has_builtin(__builtin_amdgcn_tanh_f32)
  return __builtin_amdgcn_tanh_f32(v);
#else
  return tanhf(v);
#endif
}

// -------------------------------------------------------------------------
// Zero the accumulation scratch (y2 [NB*128] ++ cnt [NB], contiguous).
// -------------------------------------------------------------------------
__global__ void zero_kernel(float* __restrict__ p, int n) {
  int i = blockIdx.x * blockDim.x + threadIdx.x;
  if (i < n) p[i] = 0.0f;
}

// -------------------------------------------------------------------------
// Repack a 128x128 row-major f32 weight matrix W[k][n] into f16 WMMA
// B-fragment-major layout for v_wmma_f32_16x16x32_f16 (wave32).
// B layout (ISA 7.12.2): lane half h=lane/16, N=lane&15;
//                        elem j<8: K=8h+j; j>=8: K=16+8h+(j-8).
// Fragment f = ktile*8 + ntile; linear idx = (f*32 + lane)*16 + j.
// -------------------------------------------------------------------------
__global__ void prep_weights_kernel(const float* __restrict__ W,
                                    _Float16* __restrict__ out) {
  int tid = blockIdx.x * blockDim.x + threadIdx.x;
  if (tid >= DIM * DIM) return;
  int f    = tid >> 9;         // /512
  int lane = (tid >> 4) & 31;
  int j    = tid & 15;
  int kt = f >> 3;             // ktile 0..3
  int nt = f & 7;              // ntile 0..7
  int k = kt * 32 + ((lane < 16) ? 0 : 8) + (j & 7) + ((j & 8) ? 16 : 0);
  int n = nt * 16 + (lane & 15);
  out[tid] = (_Float16)W[k * DIM + n];
}

// -------------------------------------------------------------------------
// Main per-element kernel: one wave32 handles one 16-row tile.
//   T  = X @ vW1 + vb1            (WMMA f16 -> f32)
//   h  = tanh(layernorm(T))       (register shuffles, V_TANH)
//   Y1 = h @ vW2 + vb2            (LDS transpose, WMMA)
//   segment-sum Y1 into y2[batch] (sorted runs -> few atomics)
// -------------------------------------------------------------------------
__global__ __launch_bounds__(256) void deepset_elem_kernel(
    const float* __restrict__ x, const int* __restrict__ batch,
    const float* __restrict__ vb1, const float* __restrict__ vg,
    const float* __restrict__ vbeta, const float* __restrict__ vb2,
    const _Float16* __restrict__ wB1, const _Float16* __restrict__ wB2,
    float* __restrict__ y2, float* __restrict__ cnt,
    int N, int ntiles) {
  __shared__ _Float16 __align__(16) hbuf[8][16 * DIM];  // 32 KB, 4KB/wave

  const int lane = threadIdx.x & 31;
  const int wave = threadIdx.x >> 5;
  const int tile = blockIdx.x * 8 + wave;
  if (tile >= ntiles) return;

  const int m16  = lane & 15;
  const int hi   = lane >> 4;     // half-wave select
  const int row0 = tile * 16;

  // ---- A fragments of X (f32 -> f16), ISA 16-bit A 16x32 layout:
  //      lane holds row M=lane&15; elems 0..7 -> K = kb+8*hi+j,
  //      elems 8..15 -> K = kb+16+8*hi+(j-8).
  v16h afrag[4];
  {
    const int arow   = row0 + m16;
    const bool valid = (arow < N);
    const float* xr  = x + (size_t)arow * DIM;
    #pragma unroll
    for (int c = 0; c < 4; ++c) {
      v16h a;
      if (valid) {
        const int k0 = c * 32 + 8 * hi;
        f32x4 p0 = *(const f32x4*)(xr + k0);
        f32x4 p1 = *(const f32x4*)(xr + k0 + 4);
        f32x4 p2 = *(const f32x4*)(xr + k0 + 16);
        f32x4 p3 = *(const f32x4*)(xr + k0 + 20);
        #pragma unroll
        for (int j = 0; j < 4; ++j) {
          a[j]      = (_Float16)p0[j];
          a[4 + j]  = (_Float16)p1[j];
          a[8 + j]  = (_Float16)p2[j];
          a[12 + j] = (_Float16)p3[j];
        }
      } else {
        #pragma unroll
        for (int j = 0; j < 16; ++j) a[j] = (_Float16)0.0f;
      }
      afrag[c] = a;
    }
  }

  // ---- T = X @ vW1 : 8 N-tiles x 4 K-steps = 32 WMMAs
  v8f acc[8];
  #pragma unroll
  for (int t = 0; t < 8; ++t) {
    v8f c = {0.f, 0.f, 0.f, 0.f, 0.f, 0.f, 0.f, 0.f};
    #pragma unroll
    for (int kc = 0; kc < 4; ++kc) {
      v16h b = ((const v16h*)wB1)[(kc * 8 + t) * 32 + lane];
      c = __builtin_amdgcn_wmma_f32_16x16x32_f16(false, afrag[kc], false, b,
                                                 (short)0, c, false, false);
    }
    acc[t] = c;
  }

  // ---- bias + per-row mean/var. C/D layout: elem r of acc[t] is
  //      row M = r + 8*hi, col = t*16 + m16. Rows live in a half-wave, so
  //      xor-shuffles over masks 1,2,4,8 reduce exactly the right lanes.
  float s[8], q[8];
  #pragma unroll
  for (int r = 0; r < 8; ++r) { s[r] = 0.f; q[r] = 0.f; }
  #pragma unroll
  for (int t = 0; t < 8; ++t) {
    float b1 = vb1[t * 16 + m16];
    #pragma unroll
    for (int r = 0; r < 8; ++r) {
      float v = acc[t][r] + b1;
      acc[t][r] = v;
      s[r] += v;
      q[r] += v * v;
    }
  }
  #pragma unroll
  for (int r = 0; r < 8; ++r) {
    #pragma unroll
    for (int msk = 1; msk < 16; msk <<= 1) {
      s[r] += __shfl_xor(s[r], msk, 32);
      q[r] += __shfl_xor(q[r], msk, 32);
    }
  }
  float mu[8], rs[8];
  #pragma unroll
  for (int r = 0; r < 8; ++r) {
    mu[r] = s[r] * (1.0f / 128.0f);
    float var = q[r] * (1.0f / 128.0f) - mu[r] * mu[r];
    rs[r] = rsqrtf(var + LN_EPS);
  }

  // ---- h = tanh(LN(T)) -> LDS (f16), XOR-swizzled in 16B chunks so the
  //      A-fragment ds_read_b128s below are bank-conflict-free.
  _Float16* hb = &hbuf[wave][0];
  #pragma unroll
  for (int t = 0; t < 8; ++t) {
    int col = t * 16 + m16;
    float g  = vg[col];
    float be = vbeta[col];
    int cb = col >> 3, off = col & 7;
    #pragma unroll
    for (int r = 0; r < 8; ++r) {
      int M = r + 8 * hi;
      float h = fast_tanh((acc[t][r] - mu[r]) * rs[r] * g + be);
      hb[M * DIM + (((cb ^ M) & 15) << 3) + off] = (_Float16)h;
    }
  }
  __asm volatile("s_wait_dscnt 0x0" ::: "memory");  // same-wave LDS RAW fence

  // ---- A fragments of h from LDS (row M = m16, swizzled chunks)
  v16h a2[4];
  {
    const v8h* hv = (const v8h*)hb;
    #pragma unroll
    for (int c = 0; c < 4; ++c) {
      int cb0 = (c * 32 + 8 * hi) >> 3;        // first 16B chunk
      int cb1 = (c * 32 + 16 + 8 * hi) >> 3;   // second 16B chunk
      v8h lo = hv[m16 * (DIM / 8) + ((cb0 ^ m16) & 15)];
      v8h hh = hv[m16 * (DIM / 8) + ((cb1 ^ m16) & 15)];
      v16h a;
      #pragma unroll
      for (int j = 0; j < 8; ++j) { a[j] = lo[j]; a[8 + j] = hh[j]; }
      a2[c] = a;
    }
  }

  // ---- Y1 = h @ vW2 + vb2
  v8f acc2[8];
  #pragma unroll
  for (int t = 0; t < 8; ++t) {
    v8f c = {0.f, 0.f, 0.f, 0.f, 0.f, 0.f, 0.f, 0.f};
    #pragma unroll
    for (int kc = 0; kc < 4; ++kc) {
      v16h b = ((const v16h*)wB2)[(kc * 8 + t) * 32 + lane];
      c = __builtin_amdgcn_wmma_f32_16x16x32_f16(false, a2[kc], false, b,
                                                 (short)0, c, false, false);
    }
    float b2 = vb2[t * 16 + m16];
    #pragma unroll
    for (int r = 0; r < 8; ++r) c[r] += b2;
    acc2[t] = c;
  }

  // ---- sorted-run segment reduction: one atomic set per distinct batch id
  int bid = 0x7FFFFFFF;  // sentinel for OOB rows / upper half lanes
  if (lane < 16 && (row0 + lane) < N) bid = batch[row0 + lane];

  int i = 0;
  while (i < 16) {
    int bv = __shfl(bid, i, 32);
    if (bv == 0x7FFFFFFF) break;
    unsigned long long mb = __ballot(lane < 16 && bid == bv);
    unsigned mlow = (unsigned)(mb & 0xFFFFull);
    int last = 31 - __builtin_clz(mlow);   // run is contiguous (sorted)
    #pragma unroll
    for (int t = 0; t < 8; ++t) {
      float p = 0.f;
      #pragma unroll
      for (int r = 0; r < 8; ++r) {
        int idx = r + 8 * hi;
        if (idx >= i && idx <= last) p += acc2[t][r];
      }
      p += __shfl_xor(p, 16, 32);  // lane and lane^16 share the same column
      if (hi == 0) atomicAdd(&y2[(size_t)bv * DIM + t * 16 + m16], p);
    }
    if (lane == 0) atomicAdd(&cnt[bv], (float)(last - i + 1));
    i = last + 1;
  }
}

// -------------------------------------------------------------------------
// Encoder: z = tanh(y2 @ eW1 + eb1) @ eW2 + eb2 ; out = [n | z]  (tiny)
// -------------------------------------------------------------------------
__global__ __launch_bounds__(64) void encoder_kernel(
    const float* __restrict__ y2, const float* __restrict__ cnt,
    const float* __restrict__ eW1, const float* __restrict__ eb1,
    const float* __restrict__ eW2, const float* __restrict__ eb2,
    float* __restrict__ out) {
  __shared__ float sy[DIM];
  __shared__ float st[MID_E];
  const int b = blockIdx.x;
  const int tid = threadIdx.x;

  sy[tid]      = y2[(size_t)b * DIM + tid];
  sy[tid + 64] = y2[(size_t)b * DIM + tid + 64];
  __syncthreads();

  for (int j = tid; j < MID_E; j += 64) {
    float a = eb1[j];
    #pragma unroll 8
    for (int k = 0; k < DIM; ++k) a += sy[k] * eW1[k * MID_E + j];
    st[j] = fast_tanh(a);
  }
  __syncthreads();

  float a = eb2[tid];
  #pragma unroll 8
  for (int j = 0; j < MID_E; ++j) a += st[j] * eW2[j * HID + tid];
  out[(size_t)b * (HID + 1) + 1 + tid] = a;
  if (tid == 0) out[(size_t)b * (HID + 1)] = cnt[b];
}

// -------------------------------------------------------------------------
extern "C" void kernel_launch(void* const* d_in, const int* in_sizes, int n_in,
                              void* d_out, int out_size, void* d_ws, size_t ws_size,
                              hipStream_t stream) {
  const float* x     = (const float*)d_in[0];
  const int*   batch = (const int*)d_in[1];
  // d_in[2] = n_batches (device scalar, derived from out_size instead)
  const float* vW1   = (const float*)d_in[3];
  const float* vb1   = (const float*)d_in[4];
  const float* vg    = (const float*)d_in[5];
  const float* vbeta = (const float*)d_in[6];
  const float* vW2   = (const float*)d_in[7];
  const float* vb2   = (const float*)d_in[8];
  const float* eW1   = (const float*)d_in[9];
  const float* eb1   = (const float*)d_in[10];
  const float* eW2   = (const float*)d_in[11];
  const float* eb2   = (const float*)d_in[12];

  const int N  = in_sizes[0] / DIM;
  const int NB = out_size / (HID + 1);

  // workspace: y2 [NB*128] | cnt [NB] | wB1 f16 [16384] | wB2 f16 [16384]
  float* y2v  = (float*)d_ws;
  float* cntv = y2v + (size_t)NB * DIM;
  _Float16* wB1 = (_Float16*)(cntv + NB);
  _Float16* wB2 = wB1 + DIM * DIM;

  const int zn = NB * (DIM + 1);
  zero_kernel<<<(zn + 255) / 256, 256, 0, stream>>>(y2v, zn);
  prep_weights_kernel<<<(DIM * DIM + 255) / 256, 256, 0, stream>>>(vW1, wB1);
  prep_weights_kernel<<<(DIM * DIM + 255) / 256, 256, 0, stream>>>(vW2, wB2);

  const int ntiles = (N + 15) / 16;
  deepset_elem_kernel<<<(ntiles + 7) / 8, 256, 0, stream>>>(
      x, batch, vb1, vg, vbeta, vb2, wB1, wB2, y2v, cntv, N, ntiles);

  encoder_kernel<<<NB, 64, 0, stream>>>(y2v, cntv, eW1, eb1, eW2, eb2,
                                        (float*)d_out);
}
